// TransBlock_5669356830940
// MI455X (gfx1250) — compile-verified, matrix operand order checked by reading, not run on hardware
//
#include <hip/hip_runtime.h>
#include <hip/hip_bf16.h>

typedef __attribute__((ext_vector_type(16))) _Float16 v16h;
typedef __attribute__((ext_vector_type(8)))  _Float16 v8h;
typedef __attribute__((ext_vector_type(8)))  float    v8f;

static constexpr int BBc  = 4;
static constexpr int CCc  = 256;
static constexpr int HHc  = 96;
static constexpr int WWc  = 96;
static constexpr int NNc  = HHc * WWc;      // 9216
static constexpr int MMc  = BBc * NNc;      // 36864
static constexpr int HIDc = 4 * CCc;        // 1024
static constexpr int HDc  = 32;             // head dim
static constexpr int APAD = 72;             // LDS row pitch (halves), 144B = 16B-aligned
static constexpr int BPAD = 72;

// ---------------------------------------------------------------- reductions
__device__ __forceinline__ float block_reduce_sum256(float v, float* sm) {
  #pragma unroll
  for (int off = 16; off > 0; off >>= 1) v += __shfl_xor(v, off, 32);
  const int w = threadIdx.x >> 5;
  __syncthreads();
  if ((threadIdx.x & 31) == 0) sm[w] = v;
  __syncthreads();
  float r = sm[0];
  #pragma unroll
  for (int i = 1; i < 8; ++i) r += sm[i];
  return r;
}

// ------------------------------------------------------- LayerNorm (2 forms)
// x: f32 [B,C,N]; out: f16 [B*N, C]. Thread-per-n: loads are n-contiguous
// (coalesced 128B per c step); 2nd pass re-reads from L2 (tensor fits in 192MB).
__global__ __launch_bounds__(256)
void ln_nchw_f16_kernel(const float* __restrict__ x, const float* __restrict__ g,
                        const float* __restrict__ bt, _Float16* __restrict__ out) {
  const int tile = blockIdx.x;                    // B * (N/256)
  const int b = tile / (NNc / 256);
  const int n = (tile % (NNc / 256)) * 256 + threadIdx.x;
  const float* xb = x + (size_t)b * CCc * NNc + n;
  float s = 0.f, s2 = 0.f;
  for (int c = 0; c < CCc; ++c) {
    float v = xb[(size_t)c * NNc];
    s += v; s2 += v * v;
  }
  const float mu = s * (1.0f / CCc);
  const float var = fmaxf(s2 * (1.0f / CCc) - mu * mu, 0.0f);
  const float rstd = rsqrtf(var + 1e-5f);
  _Float16* orow = out + ((size_t)b * NNc + n) * CCc;
  for (int c0 = 0; c0 < CCc; c0 += 8) {
    v8h pk;
    #pragma unroll
    for (int j = 0; j < 8; ++j) {
      const int c = c0 + j;
      float v = xb[(size_t)c * NNc];
      pk[j] = (_Float16)((v - mu) * rstd * g[c] + bt[c]);
    }
    *(v8h*)&orow[c0] = pk;                        // 16B store per thread
  }
}

// x: f16 [M, C] row-major, out: f16 [M, C]; one block per row (coalesced)
__global__ __launch_bounds__(256)
void ln_rows_f16_kernel(const _Float16* __restrict__ x, const float* __restrict__ g,
                        const float* __restrict__ bt, _Float16* __restrict__ out) {
  __shared__ float sm[8];
  const int bn = blockIdx.x;
  const int c = threadIdx.x;
  float v = (float)x[(size_t)bn * CCc + c];
  float mu = block_reduce_sum256(v, sm) * (1.0f / CCc);
  float d = v - mu;
  float var = block_reduce_sum256(d * d, sm) * (1.0f / CCc);
  float rstd = rsqrtf(var + 1e-5f);
  out[(size_t)bn * CCc + c] = (_Float16)(d * rstd * g[c] + bt[c]);
}

// ---------------------------------------------------------------- f32 -> f16
__global__ void cast_f32_f16_kernel(const float* __restrict__ in,
                                    _Float16* __restrict__ out, int n) {
  int i = blockIdx.x * blockDim.x + threadIdx.x;
  if (i < n) out[i] = (_Float16)in[i];
}

// ----------------------------------------------------------- WMMA f16 GEMM
// out[m,o] = act( sum_k A[m,k]*W[o,k] + bias[o] );  A:[M,K] f16, W:[O,K] f16.
// Block tile 128(M)x64(O); K consumed in 64-wide slabs, async-copied into
// double-buffered LDS (global_load_async_to_lds_b128 / ASYNCcnt), fragments
// gathered from LDS with ds_load_b128 per the ISA 16-bit A/B layouts.
__device__ __forceinline__ void async_cp16(const _Float16* g, const _Float16* l) {
  const unsigned lds = (unsigned)(uintptr_t)l;    // aperture low 32 bits = LDS offset
  asm volatile("global_load_async_to_lds_b128 %0, %1, off"
               :: "v"(lds), "v"(g) : "memory");
}

__device__ __forceinline__ void issue_slab(const _Float16* __restrict__ A,
                                           const _Float16* __restrict__ Wt,
                                           int K, int mBase, int oBase, int k0,
                                           _Float16* Al, _Float16* Bl, int t) {
  #pragma unroll
  for (int i = 0; i < 4; ++i) {                   // A: 128 rows x 64 halves
    const int chunk = t + i * 256;                // 1024 x 16B chunks
    const int r = chunk >> 3, cc = (chunk & 7) * 8;
    async_cp16(A + (size_t)(mBase + r) * K + k0 + cc, Al + r * APAD + cc);
  }
  #pragma unroll
  for (int i = 0; i < 2; ++i) {                   // B: 64 rows x 64 halves
    const int chunk = t + i * 256;                // 512 x 16B chunks
    const int r = chunk >> 3, cc = (chunk & 7) * 8;
    async_cp16(Wt + (size_t)(oBase + r) * K + k0 + cc, Bl + r * BPAD + cc);
  }
}

// lane<16: row row0+sub, K {kk..kk+7, kk+16..kk+23}; lane>=16: K {+8.., +24..}
__device__ __forceinline__ v16h lds_frag(const _Float16* base, int pad,
                                         int row0, int sub, int hi, int kk) {
  const _Float16* p = base + (row0 + sub) * pad + kk + hi * 8;
  v16h r;
  ((v8h*)&r)[0] = *(const v8h*)p;
  ((v8h*)&r)[1] = *(const v8h*)(p + 16);
  return r;
}

__global__ __launch_bounds__(256)
void wmma_gemm_kernel(const _Float16* __restrict__ A,
                      const _Float16* __restrict__ Wt,
                      const float* __restrict__ bias,
                      _Float16* __restrict__ outH,
                      float* __restrict__ outF,
                      int K, int O, int act, int storeT) {
  __shared__ __align__(16) _Float16 Al[2][128 * APAD];
  __shared__ __align__(16) _Float16 Bl[2][64 * BPAD];
  const int t = threadIdx.x;
  const int lane = t & 31, wv = t >> 5;
  const int sub = lane & 15, hi = lane >> 4;
  const int mBase = blockIdx.y * 128;
  const int oBase = blockIdx.x * 64;
  const int S = K >> 6;                           // 64-wide K slabs

  v8f acc[4] = {{}, {}, {}, {}};
  issue_slab(A, Wt, K, mBase, oBase, 0, Al[0], Bl[0], t);

  for (int s = 0; s < S; ++s) {
    const int cur = s & 1;
    if (s + 1 < S) {
      issue_slab(A, Wt, K, mBase, oBase, (s + 1) << 6, Al[1 - cur], Bl[1 - cur], t);
      asm volatile("s_wait_asynccnt 0x6" ::: "memory");  // slab s done, s+1 in flight
    } else {
      asm volatile("s_wait_asynccnt 0x0" ::: "memory");
    }
    __syncthreads();                              // all waves' portions visible
    #pragma unroll
    for (int kk = 0; kk < 64; kk += 32) {
      v16h a = lds_frag(Al[cur], APAD, wv * 16, sub, hi, kk);
      v16h bf[4];
      #pragma unroll
      for (int j = 0; j < 4; ++j)
        bf[j] = lds_frag(Bl[cur], BPAD, j * 16, sub, hi, kk);
      acc[0] = __builtin_amdgcn_wmma_f32_16x16x32_f16(false, a, false, bf[0],
                                                      (short)0, acc[0], false, false);
      acc[1] = __builtin_amdgcn_wmma_f32_16x16x32_f16(false, a, false, bf[1],
                                                      (short)0, acc[1], false, false);
      acc[2] = __builtin_amdgcn_wmma_f32_16x16x32_f16(false, a, false, bf[2],
                                                      (short)0, acc[2], false, false);
      acc[3] = __builtin_amdgcn_wmma_f32_16x16x32_f16(false, a, false, bf[3],
                                                      (short)0, acc[3], false, false);
#if defined(__has_builtin)
#if __has_builtin(__builtin_amdgcn_sched_group_barrier)
      // pipeline shape per kk-step: 10 DS reads first, then the 4 matrix ops
      __builtin_amdgcn_sched_group_barrier(0x100, 10, 0);   // DS read group
      __builtin_amdgcn_sched_group_barrier(0x008, 4, 0);    // WMMA group
#endif
#endif
    }
    if (s + 1 < S) __syncthreads();               // done reading before overwrite
  }

  // C/D layout: lane<16 -> col o0+sub, VGPR r = row r; lane>=16 -> row r+8.
  const int m0 = mBase + wv * 16;
  #pragma unroll
  for (int j = 0; j < 4; ++j) {
    const int o = oBase + j * 16 + sub;
    const float bv = bias ? bias[o] : 0.0f;
    #pragma unroll
    for (int r = 0; r < 8; ++r) {
      const int m = m0 + r + hi * 8;
      float v = acc[j][r] + bv;
      if (act == 1) v = (v > 0.0f) ? v : (__expf(v) - 1.0f);   // ELU
      if (storeT == 0) {
        outH[(size_t)m * O + o] = (_Float16)v;
      } else {                                    // f32, transposed to [B,C,H,W]
        const int b = m / NNc, n = m - b * NNc;
        outF[((size_t)b * O + o) * NNc + n] = v;
      }
    }
  }
}

// ------------------------------------------------ per-(b,channel) l2norm / N
// One block = (b, 32-channel group). A wave covers 32 consecutive channels of
// one n-row -> contiguous 64B accesses. 8 n-strided groups reduce via LDS.
__global__ __launch_bounds__(256)
void l2norm_col_kernel(_Float16* __restrict__ x, int ld) {
  __shared__ float sm[256];
  __shared__ float inv[32];
  const int b = blockIdx.x >> 3;
  const int c = ((blockIdx.x & 7) << 5) + (threadIdx.x & 31);
  const int g = threadIdx.x >> 5;
  _Float16* base = x + (size_t)b * NNc * ld + c;
  float s = 0.0f;
  for (int n = g; n < NNc; n += 8) {
    float v = (float)base[(size_t)n * ld];
    s += v * v;
  }
  sm[threadIdx.x] = s;
  __syncthreads();
  if (g == 0) {
    float tot = 0.0f;
    #pragma unroll
    for (int i = 0; i < 8; ++i) tot += sm[i * 32 + (threadIdx.x & 31)];
    inv[threadIdx.x] = 1.0f / fmaxf(sqrtf(tot), 1e-12f);
  }
  __syncthreads();
  const float iv = inv[threadIdx.x & 31];
  for (int n = g; n < NNc; n += 8) {
    base[(size_t)n * ld] = (_Float16)((float)base[(size_t)n * ld] * iv);
  }
}

// --------------------------------------- cross-covariance attention 32x32/hd
__global__ __launch_bounds__(256)
void attn_kernel(const _Float16* __restrict__ q, const _Float16* __restrict__ kv,
                 _Float16* __restrict__ attn) {
  __shared__ __align__(16) _Float16 qs[128 * 32];
  __shared__ __align__(16) _Float16 ks[128 * 32];
  __shared__ float S[32 * 33];
  const int bh = blockIdx.x;
  const int b = bh >> 3, h = bh & 7;
  const int t = threadIdx.x;
  const int d = t >> 3;
  const int e0 = (t & 7) * 4;
  float acc[4] = {0.f, 0.f, 0.f, 0.f};

  for (int n0 = 0; n0 < NNc; n0 += 128) {
    __syncthreads();
    #pragma unroll
    for (int p = 0; p < 2; ++p) {
      const int r = p * 64 + (t >> 2);
      const int col8 = (t & 3) * 8;
      const size_t nrow = (size_t)b * NNc + n0 + r;
      *(v8h*)&qs[r * 32 + col8] = *(const v8h*)(q  + nrow * CCc       + h * HDc + col8);
      *(v8h*)&ks[r * 32 + col8] = *(const v8h*)(kv + nrow * (2 * CCc) + h * HDc + col8);
    }
    __syncthreads();
    #pragma unroll 4
    for (int nn = 0; nn < 128; ++nn) {
      const float qv = (float)qs[nn * 32 + d];
      acc[0] += qv * (float)ks[nn * 32 + e0 + 0];
      acc[1] += qv * (float)ks[nn * 32 + e0 + 1];
      acc[2] += qv * (float)ks[nn * 32 + e0 + 2];
      acc[3] += qv * (float)ks[nn * 32 + e0 + 3];
    }
  }
  S[d * 33 + e0 + 0] = acc[0];
  S[d * 33 + e0 + 1] = acc[1];
  S[d * 33 + e0 + 2] = acc[2];
  S[d * 33 + e0 + 3] = acc[3];
  __syncthreads();
  if (t < 32) {
    const float scale = 0.17677669529663687f;     // 32^-0.5
    float mx = -1e30f;
    for (int e = 0; e < 32; ++e) mx = fmaxf(mx, S[t * 33 + e] * scale);
    float p[32], sum = 0.0f;
    for (int e = 0; e < 32; ++e) { p[e] = __expf(S[t * 33 + e] * scale - mx); sum += p[e]; }
    const float inv = 1.0f / sum;
    for (int e = 0; e < 32; ++e)
      attn[((size_t)bh * 32 + t) * 32 + e] = (_Float16)(p[e] * inv);
  }
}

// out[m, h*32+d] = sum_e attn[b,h,d,e] * v[m, 256 + h*32+e]
__global__ __launch_bounds__(256)
void attnv_kernel(const _Float16* __restrict__ attn, const _Float16* __restrict__ kv,
                  _Float16* __restrict__ out) {
  const int m = blockIdx.x;
  const int c = threadIdx.x;
  const int b = m / NNc;
  const int h = c >> 5, d = c & 31;
  const _Float16* arow = attn + ((size_t)((b * 8 + h) * 32 + d)) * 32;
  const _Float16* vrow = kv + (size_t)m * (2 * CCc) + CCc + h * HDc;
  float acc = 0.0f;
  #pragma unroll
  for (int e = 0; e < 32; ++e) acc += (float)arow[e] * (float)vrow[e];
  out[(size_t)m * CCc + c] = (_Float16)acc;
}

// ---------------------------------------- depthwise 3x3 conv, reflect pad 1
__global__ __launch_bounds__(256)
void dwconv3x3_kernel(const _Float16* __restrict__ x, const float* __restrict__ cw,
                      const float* __restrict__ cb, _Float16* __restrict__ out) {
  const int m = blockIdx.x;
  const int c = threadIdx.x;
  const int b = m / NNc, n = m - b * NNc;
  const int h = n / WWc, w = n - h * WWc;
  float acc = cb[c];
  #pragma unroll
  for (int i = 0; i < 3; ++i) {
    int rh = h + i - 1;
    rh = rh < 0 ? -rh : (rh >= HHc ? 2 * HHc - 2 - rh : rh);
    #pragma unroll
    for (int j = 0; j < 3; ++j) {
      int rw = w + j - 1;
      rw = rw < 0 ? -rw : (rw >= WWc ? 2 * WWc - 2 - rw : rw);
      acc += cw[c * 9 + i * 3 + j] *
             (float)x[((size_t)(b * NNc + rh * WWc + rw)) * CCc + c];
    }
  }
  out[(size_t)m * CCc + c] = (_Float16)acc;
}

// ---------------------------------------------------------------- launcher
extern "C" void kernel_launch(void* const* d_in, const int* in_sizes, int n_in,
                              void* d_out, int out_size, void* d_ws, size_t ws_size,
                              hipStream_t stream) {
  (void)in_sizes; (void)n_in; (void)out_size; (void)ws_size;
  const float* ctx_x  = (const float*)d_in[0];
  const float* dep_x  = (const float*)d_in[1];
  const float* nctx_w = (const float*)d_in[2];
  const float* nctx_b = (const float*)d_in[3];
  const float* ndep_w = (const float*)d_in[4];
  const float* ndep_b = (const float*)d_in[5];
  const float* q_w    = (const float*)d_in[6];
  const float* kv_w   = (const float*)d_in[7];
  const float* proj_w = (const float*)d_in[8];
  const float* proj_b = (const float*)d_in[9];
  const float* nmlp_w = (const float*)d_in[10];
  const float* nmlp_b = (const float*)d_in[11];
  const float* conv_w = (const float*)d_in[12];
  const float* conv_b = (const float*)d_in[13];
  const float* mlpn_w = (const float*)d_in[14];
  const float* mlpn_b = (const float*)d_in[15];
  const float* fc1_w  = (const float*)d_in[16];
  const float* fc1_b  = (const float*)d_in[17];
  const float* fc2_w  = (const float*)d_in[18];
  const float* fc2_b  = (const float*)d_in[19];
  float* outp = (float*)d_out;

  char* ws = (char*)d_ws;
  size_t off = 0;
  auto take = [&](size_t bytes) -> char* {
    char* p = ws + off;
    off += (bytes + 255) & ~(size_t)255;
    return p;
  };
  _Float16* buf0   = (_Float16*)take((size_t)MMc * CCc * 2);
  _Float16* buf1   = (_Float16*)take((size_t)MMc * CCc * 2);
  _Float16* bufQ   = (_Float16*)take((size_t)MMc * CCc * 2);
  _Float16* bufKV  = (_Float16*)take((size_t)MMc * 2 * CCc * 2);
  _Float16* bufH   = (_Float16*)take((size_t)MMc * HIDc * 2);
  _Float16* wq     = (_Float16*)take((size_t)CCc * CCc * 2);
  _Float16* wkv    = (_Float16*)take((size_t)2 * CCc * CCc * 2);
  _Float16* wproj  = (_Float16*)take((size_t)CCc * CCc * 2);
  _Float16* wfc1   = (_Float16*)take((size_t)HIDc * CCc * 2);
  _Float16* wfc2   = (_Float16*)take((size_t)CCc * HIDc * 2);
  _Float16* bufAtt = (_Float16*)take((size_t)BBc * 8 * 32 * 32 * 2);

  const dim3 blk(256);
  // 1) weights -> f16
  cast_f32_f16_kernel<<<(CCc * CCc + 255) / 256, blk, 0, stream>>>(q_w, wq, CCc * CCc);
  cast_f32_f16_kernel<<<(2 * CCc * CCc + 255) / 256, blk, 0, stream>>>(kv_w, wkv, 2 * CCc * CCc);
  cast_f32_f16_kernel<<<(CCc * CCc + 255) / 256, blk, 0, stream>>>(proj_w, wproj, CCc * CCc);
  cast_f32_f16_kernel<<<(HIDc * CCc + 255) / 256, blk, 0, stream>>>(fc1_w, wfc1, HIDc * CCc);
  cast_f32_f16_kernel<<<(CCc * HIDc + 255) / 256, blk, 0, stream>>>(fc2_w, wfc2, CCc * HIDc);

  // 2) LayerNorms (NCHW -> [M,C] f16), coalesced over n
  ln_nchw_f16_kernel<<<BBc * (NNc / 256), blk, 0, stream>>>(ctx_x, nctx_w, nctx_b, buf0);
  ln_nchw_f16_kernel<<<BBc * (NNc / 256), blk, 0, stream>>>(dep_x, ndep_w, ndep_b, buf1);

  // 3) q / kv projections (WMMA, async-LDS pipelined)
  wmma_gemm_kernel<<<dim3(CCc / 64, MMc / 128), blk, 0, stream>>>(
      buf0, wq, nullptr, bufQ, nullptr, CCc, CCc, 0, 0);
  wmma_gemm_kernel<<<dim3(2 * CCc / 64, MMc / 128), blk, 0, stream>>>(
      buf1, wkv, nullptr, bufKV, nullptr, CCc, 2 * CCc, 0, 0);

  // 4) l2-normalize q and k along N (k = first 256 cols of kv)
  l2norm_col_kernel<<<BBc * 8, blk, 0, stream>>>(bufQ, CCc);
  l2norm_col_kernel<<<BBc * 8, blk, 0, stream>>>(bufKV, 2 * CCc);

  // 5) attention + attn @ v
  attn_kernel<<<BBc * 8, blk, 0, stream>>>(bufQ, bufKV, bufAtt);
  attnv_kernel<<<MMc, blk, 0, stream>>>(bufAtt, bufKV, buf0);

  // 6) proj (+bias), LN, depthwise conv (+bias), LN
  wmma_gemm_kernel<<<dim3(CCc / 64, MMc / 128), blk, 0, stream>>>(
      buf0, wproj, proj_b, buf1, nullptr, CCc, CCc, 0, 0);
  ln_rows_f16_kernel<<<MMc, blk, 0, stream>>>(buf1, nmlp_w, nmlp_b, buf0);
  dwconv3x3_kernel<<<MMc, blk, 0, stream>>>(buf0, conv_w, conv_b, buf1);
  ln_rows_f16_kernel<<<MMc, blk, 0, stream>>>(buf1, mlpn_w, mlpn_b, buf0);

  // 7) MLP: fc1 (+bias, ELU), fc2 (+bias, transposed f32 store into d_out)
  wmma_gemm_kernel<<<dim3(HIDc / 64, MMc / 128), blk, 0, stream>>>(
      buf0, wfc1, fc1_b, bufH, nullptr, CCc, HIDc, 1, 0);
  wmma_gemm_kernel<<<dim3(CCc / 64, MMc / 128), blk, 0, stream>>>(
      bufH, wfc2, fc2_b, nullptr, outp, HIDc, CCc, 0, 1);
}
